// DynamicGNNPBPK_62740882260651
// MI455X (gfx1250) — matrix operand order repaired and checked by reading, not compiled, not run on hardware
//
#include <hip/hip_runtime.h>
#include <hip/hip_bf16.h>
#include <math.h>

// ---------------------------------------------------------------------------
// DynamicGNNPBPK for MI455X (gfx1250), wave32 + V_WMMA_F32_16X16X32_BF16.
//
// B=8192, ORGANS=15, N=122880, K_IN=2, E=245760, HID=64, HEADS=4, THID=32, T=96.
//
// Structure exploited:
//  * col[e] = e/2 exactly -> every segment op is a fixed 2-element reduction.
//  * repeat(inp,15) @ flow == repeat(inp @ flow, 15): flow MLP runs at [B,64].
//  * (e@lin_edge_w * att_edge).sum(-1) == e @ M[3,H] -> precomputed.
//  * bandwidth-bound: big activations bf16; Linear->ReLU->Linear chains fused
//    per 16-row tile in LDS; ALL bf16 conversion hoisted out of the GEMM path
//    (weights pre-converted+padded once per launch, LDS tiles staged bf16, so
//    fragments are pure b128 loads).
//
// Input flattening assumption (jax tree flatten: dict keys sorted, lists in
// order):
//   0 dose | 1 drug_params | 2 edge_attr | 3 edge_index | 4 node_features |
//   5..67 params leaves | 68 physio_params
//   params: comb(b5,w6) drug1(7,8) drug2(9,10) edge1(11,12) edge2(13,14)
//           flow1(15,16) flow2(17,18)
//           gat[l]: base 19+6l -> att_dst,att_edge,att_src,bias,lin_edge_w,lin_w
//           lstm: b43,w_hh44,w_ih45  msg1(46,47) msg2(48,49) node1(50,51)
//           node2(52,53) out1(54,55) out2(56,57) phys1(58,59) phys2(60,61)
//           tlin(62,63) upd1(64,65) upd2(66,67)
// ---------------------------------------------------------------------------

#define B_      8192
#define ORGANS_ 15
#define N_      (B_ * ORGANS_)
#define E_      (N_ * 2)
#define T_      96

typedef __attribute__((ext_vector_type(16))) __bf16    v16bf;
typedef __attribute__((ext_vector_type(8)))  __bf16    v8bf;
typedef __attribute__((ext_vector_type(8)))  float     v8f;
typedef __attribute__((ext_vector_type(4)))  unsigned  u32x4;

__device__ __forceinline__ __bf16 f2bf(float f) {
  union { float f; unsigned u; } x; x.f = f;
  unsigned r = x.u + 0x7FFFu + ((x.u >> 16) & 1u);   // round-to-nearest-even
  unsigned short h = (unsigned short)(r >> 16);
  __bf16 b; __builtin_memcpy(&b, &h, 2); return b;
}
__device__ __forceinline__ float bf2f(__bf16 b) {
  unsigned short h; __builtin_memcpy(&h, &b, 2);
  union { unsigned u; float f; } x; x.u = ((unsigned)h) << 16; return x.f;
}

__device__ __forceinline__ v8f wmma_bf16(v16bf a, v16bf b, v8f c) {
  // emits v_wmma_f32_16x16x32_bf16 (codegen-confirmed builtin, 8 args)
  return __builtin_amdgcn_wmma_f32_16x16x32_bf16(false, a, false, b,
                                                 (short)0, c, false, false);
}

// A fragment (16x32 bf16). CDNA5 layout: lane m=lane&15; lanes 0-15 carry K
// chunks {0..7,16..23}, lanes 16-31 carry {8..15,24..31}. Per lane that is
// two contiguous 8-element (16-byte) runs -> two b128 loads + concat.
// Requires lda % 32 == 0 and 16B-aligned X (staged that way).
__device__ __forceinline__ v16bf load_a_bf(const __bf16* X, int lda, int kbase) {
  int lane = threadIdx.x & 31;
  int m  = lane & 15;
  int hi = lane >> 4;
  const __bf16* p = X + m * lda + kbase + hi * 8;
  v8bf lo = *(const v8bf*)(p);
  v8bf hv = *(const v8bf*)(p + 16);
  return __builtin_shufflevector(lo, hv, 0, 1, 2, 3, 4, 5, 6, 7,
                                         8, 9, 10, 11, 12, 13, 14, 15);
}

// B fragment (32x16 bf16): per-lane column n = ncol + (lane&15); element s has
// k = kbase + 8*(lane>>4) + s (s<8) / kbase + 16 + 8*(lane>>4) + (s-8) (s>=8).
__device__ __forceinline__ v16bf load_b_bf(const __bf16* W, int ldw, int ncol,
                                           int kbase) {
  int lane = threadIdx.x & 31;
  int n = ncol + (lane & 15);
  int hi = lane >> 4;
  v16bf b;
#pragma unroll
  for (int s = 0; s < 16; ++s) {
    int k = kbase + hi * 8 + (s < 8 ? s : s + 8);
    b[s] = W[(size_t)k * ldw + n];
  }
  return b;
}

// C/D layout: VGPR r -> row r (lanes 0-15) / r+8 (lanes 16-31), col = lane&15.
template<int ACT>  // 0 = none, 1 = relu
__device__ __forceinline__ void store_f32(float* Y, int ldy, int ncol, v8f acc,
                                          const float* bias) {
  int lane = threadIdx.x & 31;
  int n  = ncol + (lane & 15);
  int mb = (lane >> 4) << 3;
  float bv = bias ? bias[n] : 0.f;
#pragma unroll
  for (int r = 0; r < 8; ++r) {
    float v = acc[r] + bv;
    if (ACT == 1) v = fmaxf(v, 0.f);
    Y[(size_t)(mb + r) * ldy + n] = v;
  }
}
template<int ACT>
__device__ __forceinline__ void store_bf(__bf16* Y, int ldy, int ncol, v8f acc,
                                         const float* bias) {
  int lane = threadIdx.x & 31;
  int n  = ncol + (lane & 15);
  int mb = (lane >> 4) << 3;
  float bv = bias ? bias[n] : 0.f;
#pragma unroll
  for (int r = 0; r < 8; ++r) {
    float v = acc[r] + bv;
    if (ACT == 1) v = fmaxf(v, 0.f);
    Y[(size_t)(mb + r) * ldy + n] = f2bf(v);
  }
}

// Block-level fused layer: Y[16,NO] = act(X[16,K] @ W[K,NO] + b).
// K already padded to a multiple of 32 (A tile and W both zero-padded).
template<int ACT>
__device__ __forceinline__ void lds_layer(const __bf16* Xl, int lda, int K,
                                          float* Yl, int ldy, int NO,
                                          const __bf16* W, int ldw,
                                          const float* bias) {
  int wave = threadIdx.x >> 5;
  int nw   = blockDim.x >> 5;
  for (int t = wave; t < (NO >> 4); t += nw) {
    v8f acc = {0.f, 0.f, 0.f, 0.f, 0.f, 0.f, 0.f, 0.f};
    for (int k0 = 0; k0 < K; k0 += 32)
      acc = wmma_bf16(load_a_bf(Xl, lda, k0), load_b_bf(W, ldw, t * 16, k0), acc);
    store_f32<ACT>(Yl, ldy, t * 16, acc, bias);
  }
  __syncthreads();
}
template<int ACT>
__device__ __forceinline__ void lds_layer(const __bf16* Xl, int lda, int K,
                                          __bf16* Yl, int ldy, int NO,
                                          const __bf16* W, int ldw,
                                          const float* bias) {
  int wave = threadIdx.x >> 5;
  int nw   = blockDim.x >> 5;
  for (int t = wave; t < (NO >> 4); t += nw) {
    v8f acc = {0.f, 0.f, 0.f, 0.f, 0.f, 0.f, 0.f, 0.f};
    for (int k0 = 0; k0 < K; k0 += 32)
      acc = wmma_bf16(load_a_bf(Xl, lda, k0), load_b_bf(W, ldw, t * 16, k0), acc);
    store_bf<ACT>(Yl, ldy, t * 16, acc, bias);
  }
  __syncthreads();
}

// Raw bf16 tile copy, 16 bytes per thread-iteration. nelem % 8 == 0.
__device__ __forceinline__ void copy_bf_tile(__bf16* dst, const __bf16* src,
                                             int nelem) {
  u32x4* d = (u32x4*)dst;
  const u32x4* s = (const u32x4*)src;
  for (int i = threadIdx.x; i < (nelem >> 3); i += blockDim.x) d[i] = s[i];
}

// ---------------------------------------------------------------------------
// Kernel 0: weight convert f32 -> bf16 with zero padding: dst[Kp,Np] (Np pow2).
__global__ void k_cvt(const float* __restrict__ src, __bf16* __restrict__ dst,
                      int K, int Nn, int total, int npshift) {
  int i = blockIdx.x * blockDim.x + threadIdx.x;
  if (i >= total) return;
  int n = i & ((1 << npshift) - 1);
  int k = i >> npshift;
  float v = (k < K && n < Nn) ? src[k * Nn + n] : 0.f;
  dst[i] = f2bf(v);
}

// ---------------------------------------------------------------------------
// Kernel 1: per-batch front end: df/pf MLPs -> comb -> flow MLP (fused).
__global__ __launch_bounds__(128) void k_inp_flow(
    const float* __restrict__ drug, const float* __restrict__ phys,
    const __bf16* w_d1, const float* b_d1, const __bf16* w_d2, const float* b_d2,
    const __bf16* w_p1, const float* b_p1, const __bf16* w_p2, const float* b_p2,
    const __bf16* w_cb, const float* b_cb,
    const __bf16* w_f1, const float* b_f1, const __bf16* w_f2, const float* b_f2,
    float* __restrict__ ff) {
  __shared__ __align__(16) __bf16 sX[16 * 32];
  __shared__ __align__(16) __bf16 sCat[16 * 128];
  __shared__ __align__(16) __bf16 sC[16 * 64];
  __shared__ __align__(16) __bf16 sD[16 * 64];
  int row0 = blockIdx.x * 16;

  for (int i = threadIdx.x; i < 16 * 32; i += blockDim.x) {
    int r = i >> 5, c = i & 31;
    sX[i] = (c < 8) ? f2bf(drug[(size_t)(row0 + r) * 8 + c]) : f2bf(0.f);
  }
  __syncthreads();
  lds_layer<1>(sX, 32, 32, sC,   64,  64, w_d1, 64, b_d1);
  lds_layer<0>(sC, 64, 64, sCat, 128, 64, w_d2, 64, b_d2);

  for (int i = threadIdx.x; i < 16 * 32; i += blockDim.x) {
    int r = i >> 5, c = i & 31;
    sX[i] = (c < 31) ? f2bf(phys[(size_t)(row0 + r) * 31 + c]) : f2bf(0.f);
  }
  __syncthreads();
  lds_layer<1>(sX, 32, 32, sC,        64,  64, w_p1, 64, b_p1);
  lds_layer<0>(sC, 64, 64, sCat + 64, 128, 64, w_p2, 64, b_p2);

  lds_layer<1>(sCat, 128, 128, sC, 64, 64, w_cb, 64, b_cb);   // inp (relu)
  lds_layer<1>(sC,   64,  64,  sD, 64, 64, w_f1, 64, b_f1);   // flow1 relu
  lds_layer<0>(sD,   64,  64,  ff + (size_t)row0 * 64, 64, 64, w_f2, 64, b_f2);
}

// Kernel 2: node MLP + add repeated flow output -> x0 bf16 [N,64].
__global__ __launch_bounds__(128) void k_node(
    const float* __restrict__ nf,
    const __bf16* w1, const float* b1, const __bf16* w2, const float* b2,
    const float* __restrict__ ff, __bf16* __restrict__ x0) {
  __shared__ __align__(16) __bf16 sX[16 * 32];
  __shared__ __align__(16) __bf16 sC[16 * 64];
  __shared__ __align__(16) float  sY[16 * 64];
  int row0 = blockIdx.x * 16;
  for (int i = threadIdx.x; i < 16 * 32; i += blockDim.x) {
    int r = i >> 5, c = i & 31;
    sX[i] = (c < 5) ? f2bf(nf[(size_t)(row0 + r) * 5 + c]) : f2bf(0.f);
  }
  __syncthreads();
  lds_layer<1>(sX, 32, 32, sC, 64, 64, w1, 64, b1);
  lds_layer<0>(sC, 64, 64, sY, 64, 64, w2, 64, b2);
  for (int i = threadIdx.x; i < 16 * 64; i += blockDim.x) {
    int r = i >> 6, c = i & 63;
    int n = row0 + r;
    x0[(size_t)n * 64 + c] = f2bf(sY[i] + ff[(size_t)(n / ORGANS_) * 64 + c]);
  }
}

// Kernel 3: edge MLP (3->64 relu ->3), weight pre-padded to [64,16] bf16.
__global__ __launch_bounds__(128) void k_edge(
    const float* __restrict__ ea,
    const __bf16* w1, const float* b1, const __bf16* w2, const float* b2,
    float* __restrict__ edge_p) {
  __shared__ __align__(16) __bf16 sX[16 * 32];
  __shared__ __align__(16) __bf16 sC[16 * 64];
  __shared__ __align__(16) float  sY[16 * 16];
  int row0 = blockIdx.x * 16;
  for (int i = threadIdx.x; i < 16 * 32; i += blockDim.x) {
    int r = i >> 5, c = i & 31;
    sX[i] = (c < 3) ? f2bf(ea[(size_t)(row0 + r) * 3 + c]) : f2bf(0.f);
  }
  __syncthreads();
  lds_layer<1>(sX, 32, 32, sC, 64, 64, w1, 64, b1);
  lds_layer<0>(sC, 64, 64, sY, 16, 16, w2, 16, nullptr);
  for (int i = threadIdx.x; i < 16 * 3; i += blockDim.x) {
    int r = i / 3, c = i % 3;
    edge_p[(size_t)(row0 + r) * 3 + c] = sY[r * 16 + c] + b2[c];
  }
}

// Kernel 4: collapse att_edge: M[l][3][4] = lin_edge_w x att_edge.
__global__ void k_att_mat(const float* lew0, const float* ae0,
                          const float* lew1, const float* ae1,
                          const float* lew2, const float* ae2,
                          const float* lew3, const float* ae3,
                          float* __restrict__ M) {
  int t = threadIdx.x;
  if (t >= 48) return;
  int l = t / 12, j = (t % 12) / 4, h = t % 4;
  int H = (l == 3) ? 1 : 4;
  if (h >= H) { M[t] = 0.f; return; }
  const float* lew = (l == 0) ? lew0 : (l == 1) ? lew1 : (l == 2) ? lew2 : lew3;
  const float* ae  = (l == 0) ? ae0  : (l == 1) ? ae1  : (l == 2) ? ae2  : ae3;
  int HC = H * 64;
  float s = 0.f;
  for (int c = 0; c < 64; ++c) s += lew[j * HC + h * 64 + c] * ae[h * 64 + c];
  M[t] = s;
}

// Kernel 5: GAT projection xl = x @ lin_w (bf16) + a_src/a_dst reduced in LDS.
template<int KD, int HC, int H>
__global__ __launch_bounds__(256) void k_gat_proj(
    const __bf16* __restrict__ xin, const __bf16* __restrict__ lin_w,
    const float* __restrict__ att_src, const float* __restrict__ att_dst,
    __bf16* __restrict__ xl, float* __restrict__ aS, float* __restrict__ aD) {
  __shared__ __align__(16) __bf16 sX[16 * KD];
  __shared__ __align__(16) __bf16 sY[16 * HC];
  int row0 = blockIdx.x * 16;
  copy_bf_tile(sX, xin + (size_t)row0 * KD, 16 * KD);
  __syncthreads();
  lds_layer<0>(sX, KD, KD, sY, HC, HC, lin_w, HC, nullptr);
  int t = threadIdx.x;
  if (t < 16 * H) {
    int r = t / H, h = t % H;
    float ss = 0.f, sd = 0.f;
    for (int c = 0; c < 64; ++c) {
      float v = bf2f(sY[r * HC + h * 64 + c]);
      ss += v * att_src[h * 64 + c];
      sd += v * att_dst[h * 64 + c];
    }
    aS[(size_t)(row0 + r) * H + h] = ss;
    aD[(size_t)(row0 + r) * H + h] = sd;
  }
  copy_bf_tile(xl + (size_t)row0 * HC, sY, 16 * HC);
}

// Kernel 6: GAT 2-edge softmax + aggregate + bias(+elu).
template<int H, bool ELU>
__global__ __launch_bounds__(256) void k_gat_agg(
    const __bf16* __restrict__ xl, const float* __restrict__ aS,
    const float* __restrict__ aD, const float* __restrict__ edge_p,
    const int* __restrict__ rowidx, const float* __restrict__ M,
    const float* __restrict__ bias, __bf16* __restrict__ xout) {
  const int HC = H * 64;
  __shared__ float sw0[32 * H], sw1[32 * H];
  __shared__ int   sr0[32],     sr1[32];
  int node0 = blockIdx.x * 32;
  int t = threadIdx.x;
  if (t < 32 * H) {
    int nl = t / H, h = t % H;
    int n = node0 + nl;
    int e0 = 2 * n, e1 = 2 * n + 1;
    int r0 = rowidx[e0], r1 = rowidx[e1];
    if (h == 0) {
      sr0[nl] = r0; sr1[nl] = r1;
      __builtin_prefetch((const void*)(xl + (size_t)r0 * HC), 0, 1);  // global_prefetch_b8
      __builtin_prefetch((const void*)(xl + (size_t)r1 * HC), 0, 1);
    }
    float ad = aD[(size_t)n * H + h];
    float a0 = aS[(size_t)r0 * H + h] + ad +
               edge_p[(size_t)e0 * 3 + 0] * M[0 * 4 + h] +
               edge_p[(size_t)e0 * 3 + 1] * M[1 * 4 + h] +
               edge_p[(size_t)e0 * 3 + 2] * M[2 * 4 + h];
    float a1 = aS[(size_t)r1 * H + h] + ad +
               edge_p[(size_t)e1 * 3 + 0] * M[0 * 4 + h] +
               edge_p[(size_t)e1 * 3 + 1] * M[1 * 4 + h] +
               edge_p[(size_t)e1 * 3 + 2] * M[2 * 4 + h];
    a0 = (a0 > 0.f) ? a0 : 0.2f * a0;   // leaky_relu(0.2)
    a1 = (a1 > 0.f) ? a1 : 0.2f * a1;
    float m  = fmaxf(a0, a1);
    float w0 = __expf(a0 - m), w1 = __expf(a1 - m);
    float s  = w0 + w1;
    sw0[t] = w0 / s; sw1[t] = w1 / s;
  }
  __syncthreads();
  for (int i = t; i < 32 * HC; i += blockDim.x) {
    int nl = i / HC, c = i % HC;   // HC is a power of two
    int h = c >> 6;
    float v = sw0[nl * H + h] * bf2f(xl[(size_t)sr0[nl] * HC + c]) +
              sw1[nl * H + h] * bf2f(xl[(size_t)sr1[nl] * HC + c]) + bias[c];
    if (ELU) v = (v > 0.f) ? v : (__expf(v) - 1.f);
    xout[(size_t)(node0 + nl) * HC + c] = f2bf(v);
  }
}

// Kernel 7: message MLP on [x[row]|x[col]|edge_p] (K padded 131->160), then the
// 2-edge mean -> agg bf16.
__global__ __launch_bounds__(128) void k_msg(
    const __bf16* __restrict__ x, const float* __restrict__ edge_p,
    const int* __restrict__ rowidx,
    const __bf16* w1, const float* b1, const __bf16* w2, const float* b2,
    __bf16* __restrict__ agg) {
  __shared__ __align__(16) __bf16 sX[16 * 160];
  __shared__ __align__(16) __bf16 sC[16 * 64];
  __shared__ __align__(16) float  sY[16 * 64];
  int e0 = blockIdx.x * 16;
  for (int i = threadIdx.x; i < 16 * 160; i += blockDim.x) {
    int r = i / 160, c = i % 160;
    int e = e0 + r;
    __bf16 v = f2bf(0.f);
    if (c < 64)       v = x[(size_t)rowidx[e] * 64 + c];
    else if (c < 128) v = x[(size_t)(e >> 1) * 64 + (c - 64)];
    else if (c < 131) v = f2bf(edge_p[(size_t)e * 3 + (c - 128)]);
    sX[i] = v;
  }
  __syncthreads();
  lds_layer<1>(sX, 160, 160, sC, 64, 64, w1, 64, b1);
  lds_layer<0>(sC, 64, 64,   sY, 64, 64, w2, 64, b2);
  int n0 = e0 >> 1;
  for (int i = threadIdx.x; i < 8 * 64; i += blockDim.x) {
    int nl = i >> 6, c = i & 63;
    agg[(size_t)(n0 + nl) * 64 + c] =
        f2bf(0.5f * (sY[(2 * nl) * 64 + c] + sY[(2 * nl + 1) * 64 + c]));
  }
}

// Kernel 8: update MLP on [x|agg] fused with LSTM input projection.
__global__ __launch_bounds__(256) void k_upd(
    const __bf16* __restrict__ x, const __bf16* __restrict__ agg,
    const __bf16* wu1, const float* bu1, const __bf16* wu2, const float* bu2,
    const __bf16* w_ih, const float* lb, float* __restrict__ pre) {
  __shared__ __align__(16) __bf16 sX[16 * 128];
  __shared__ __align__(16) __bf16 sC[16 * 64];
  __shared__ __align__(16) __bf16 sY[16 * 64];
  int row0 = blockIdx.x * 16;
  for (int i = threadIdx.x; i < 16 * 128; i += blockDim.x) {
    int r = i >> 7, c = i & 127;
    sX[i] = (c < 64) ? x[(size_t)(row0 + r) * 64 + c]
                     : agg[(size_t)(row0 + r) * 64 + (c - 64)];
  }
  __syncthreads();
  lds_layer<1>(sX, 128, 128, sC, 64, 64, wu1, 64, bu1);
  lds_layer<0>(sC, 64, 64,   sY, 64, 64, wu2, 64, bu2);
  lds_layer<0>(sY, 64, 64, pre + (size_t)row0 * 128, 128, 128, w_ih, 128, lb);
}

// Kernel 9: serial LSTM (batch 1, 122880 dependent steps), w_hh pinned in LDS.
__global__ __launch_bounds__(128) void k_lstm(const float* __restrict__ pre,
                                              const float* __restrict__ w_hh,
                                              float* __restrict__ hs) {
  __shared__ float sW[32 * 128];
  __shared__ float h[32], c[32], z[128];
  int t = threadIdx.x;
  for (int i = t; i < 32 * 128; i += blockDim.x) sW[i] = w_hh[i];
  if (t < 32) { h[t] = 0.f; c[t] = 0.f; }
  __syncthreads();
  for (int n = 0; n < N_; ++n) {
    float acc = pre[(size_t)n * 128 + t];
#pragma unroll 8
    for (int k = 0; k < 32; ++k) acc += h[k] * sW[k * 128 + t];
    z[t] = acc;
    __syncthreads();
    if (t < 32) {
      float si = 1.f / (1.f + __expf(-z[t]));
      float sf = 1.f / (1.f + __expf(-z[32 + t]));
      float cg = tanhf(z[64 + t]);
      float so = 1.f / (1.f + __expf(-z[96 + t]));
      float cn = sf * c[t] + si * cg;
      c[t] = cn;
      float hn = so * tanhf(cn);
      h[t] = hn;
      hs[(size_t)n * 32 + t] = hn;
    }
    __syncthreads();
  }
}

// Kernel 10: head: tlin(32->64) -> out1(relu,64->32) -> out2(32->1) -> sigmoid.
__global__ __launch_bounds__(128) void k_head(
    const float* __restrict__ hs,
    const __bf16* wt, const float* bt, const __bf16* wo1, const float* bo1,
    const __bf16* wo2, const float* bo2, float* __restrict__ conc) {
  __shared__ __align__(16) __bf16 sX[16 * 32];
  __shared__ __align__(16) __bf16 sC[16 * 64];
  __shared__ __align__(16) __bf16 sD[16 * 32];
  int row0 = blockIdx.x * 16;
  for (int i = threadIdx.x; i < 16 * 32; i += blockDim.x)
    sX[i] = f2bf(hs[(size_t)row0 * 32 + i]);
  __syncthreads();
  lds_layer<0>(sX, 32, 32, sC, 64, 64, wt, 64, bt);
  lds_layer<1>(sC, 64, 64, sD, 32, 32, wo1, 32, bo1);
  if ((threadIdx.x >> 5) == 0) {           // wave-uniform: EXEC all-ones
    v8f acc = {0.f, 0.f, 0.f, 0.f, 0.f, 0.f, 0.f, 0.f};
    acc = wmma_bf16(load_a_bf(sD, 32, 0), load_b_bf(wo2, 16, 0, 0), acc);
    int lane = threadIdx.x & 31;
    if ((lane & 15) == 0) {
      int mb = (lane >> 4) << 3;
#pragma unroll
      for (int r = 0; r < 8; ++r) {
        float v = acc[r] + bo2[0];
        conc[row0 + mb + r] = 1.f / (1.f + __expf(-v));
      }
    }
  }
}

// Kernel 11: broadcast conc[N] over T steps into out[T,B,ORGANS,1].
__global__ void k_bcast(const float* __restrict__ conc, float* __restrict__ out) {
  int i = blockIdx.x * blockDim.x + threadIdx.x;
  if (i < T_ * N_) out[i] = conc[i % N_];
}

// ---------------------------------------------------------------------------
extern "C" void kernel_launch(void* const* d_in, const int* in_sizes, int n_in,
                              void* d_out, int out_size, void* d_ws, size_t ws_size,
                              hipStream_t stream) {
  (void)in_sizes; (void)n_in; (void)out_size; (void)ws_size;
#define F(i) ((const float*)d_in[i])
  const float* drug = F(1);
  const float* ea   = F(2);
  const int*   eidx = (const int*)d_in[3];   // [2,E]; row = eidx[0:E]
  const float* nf   = F(4);
  const float* phys = F(68);
  const int* rowidx = eidx;                  // dose (d_in[0]) unused, as in ref

  // Workspace layout (bytes)
  char* ws = (char*)d_ws;
  const size_t OFF_P0 = 0;                               // x    bf16 [N,256]
  const size_t OFF_P1 = OFF_P0 + (size_t)N_ * 256 * 2;   // xl bf16 / pre f32
  const size_t OFF_FF = OFF_P1 + (size_t)N_ * 256 * 2;   // ff   f32  [B,64]
  const size_t OFF_EP = OFF_FF + (size_t)B_ * 64 * 4;    // edge_p f32 [E,3]
  const size_t OFF_AS = OFF_EP + (size_t)E_ * 3 * 4;     // a_src f32 [N,4]/conc
  const size_t OFF_AD = OFF_AS + (size_t)N_ * 4 * 4;     // a_dst f32 [N,4]
  const size_t OFF_AG = OFF_AD + (size_t)N_ * 4 * 4;     // agg bf16 [N,64]/hs f32
  const size_t OFF_M  = OFF_AG + (size_t)N_ * 64 * 4;    // M [4][3][4]
  const size_t OFF_WB = OFF_M + 256;                     // bf16 weights, padded

  __bf16* P0   = (__bf16*)(ws + OFF_P0);
  __bf16* P1   = (__bf16*)(ws + OFF_P1);
  float*  ff   = (float*)(ws + OFF_FF);
  float*  ep   = (float*)(ws + OFF_EP);
  float*  aS   = (float*)(ws + OFF_AS);
  float*  aD   = (float*)(ws + OFF_AD);
  __bf16* agg  = (__bf16*)(ws + OFF_AG);
  float*  Mb   = (float*)(ws + OFF_M);
  float*  pre  = (float*)(ws + OFF_P1);   // reuse xl region after GAT
  float*  hsb  = (float*)(ws + OFF_AG);   // reuse agg region after upd
  float*  conc = (float*)(ws + OFF_AS);   // reuse a_src region

  // ---- bf16 weight conversion (padded K to %32, skinny N to 16) ----
  // order: drug1 drug2 phys1 phys2 comb flow1 flow2 node1 node2 edge1 edge2
  //        g0lin g1lin g2lin g3lin msg1 msg2 upd1 upd2 w_ih tlin out1 out2
  const int WSRC[23] = {8, 10, 59, 61, 6, 16, 18, 51, 53, 12, 14,
                        24, 30, 36, 42, 47, 49, 65, 67, 45, 63, 55, 57};
  const int WK[23]   = {8, 64, 31, 64, 128, 64, 64, 5, 64, 3, 64,
                        64, 256, 256, 256, 131, 64, 128, 64, 64, 32, 64, 32};
  const int WN[23]   = {64, 64, 64, 64, 64, 64, 64, 64, 64, 64, 3,
                        256, 256, 256, 64, 64, 64, 64, 64, 128, 64, 32, 1};
  const int WKP[23]  = {32, 64, 32, 64, 128, 64, 64, 32, 64, 32, 64,
                        64, 256, 256, 256, 160, 64, 128, 64, 64, 32, 64, 32};
  const int WNP[23]  = {64, 64, 64, 64, 64, 64, 64, 64, 64, 64, 16,
                        256, 256, 256, 64, 64, 64, 64, 64, 128, 64, 32, 16};
  __bf16* wb[23];
  {
    size_t off = 0;
    for (int i = 0; i < 23; ++i) {
      wb[i] = (__bf16*)(ws + OFF_WB) + off;
      int total = WKP[i] * WNP[i];
      int sh = 31 - __builtin_clz((unsigned)WNP[i]);   // WNP is a power of two
      k_cvt<<<(total + 255) / 256, 256, 0, stream>>>(F(WSRC[i]), wb[i],
                                                     WK[i], WN[i], total, sh);
      off += (size_t)total;
    }
  }

  k_inp_flow<<<B_ / 16, 128, 0, stream>>>(drug, phys,
      wb[0], F(7), wb[1], F(9),          // drug1/2
      wb[2], F(58), wb[3], F(60),        // phys1/2
      wb[4], F(5),                       // comb
      wb[5], F(15), wb[6], F(17), ff);   // flow1/2
  k_node<<<N_ / 16, 128, 0, stream>>>(nf, wb[7], F(50), wb[8], F(52), ff, P0);
  k_edge<<<E_ / 16, 128, 0, stream>>>(ea, wb[9], F(11), wb[10], F(13), ep);
  k_att_mat<<<1, 64, 0, stream>>>(F(23), F(20), F(29), F(26),
                                  F(35), F(32), F(41), F(38), Mb);

  // GAT stack: proj writes xl->P1, agg writes x->P0 (ping-pong stays in L2)
  k_gat_proj<64, 256, 4><<<N_ / 16, 256, 0, stream>>>(P0, wb[11], F(21), F(19), P1, aS, aD);
  k_gat_agg<4, true><<<N_ / 32, 256, 0, stream>>>(P1, aS, aD, ep, rowidx, Mb + 0,  F(22), P0);
  k_gat_proj<256, 256, 4><<<N_ / 16, 256, 0, stream>>>(P0, wb[12], F(27), F(25), P1, aS, aD);
  k_gat_agg<4, true><<<N_ / 32, 256, 0, stream>>>(P1, aS, aD, ep, rowidx, Mb + 12, F(28), P0);
  k_gat_proj<256, 256, 4><<<N_ / 16, 256, 0, stream>>>(P0, wb[13], F(33), F(31), P1, aS, aD);
  k_gat_agg<4, true><<<N_ / 32, 256, 0, stream>>>(P1, aS, aD, ep, rowidx, Mb + 24, F(34), P0);
  k_gat_proj<256, 64, 1><<<N_ / 16, 256, 0, stream>>>(P0, wb[14], F(39), F(37), P1, aS, aD);
  k_gat_agg<1, false><<<N_ / 32, 256, 0, stream>>>(P1, aS, aD, ep, rowidx, Mb + 36, F(40), P0);

  k_msg<<<E_ / 16, 128, 0, stream>>>(P0, ep, rowidx, wb[15], F(46), wb[16], F(48), agg);
  k_upd<<<N_ / 16, 256, 0, stream>>>(P0, agg, wb[17], F(64), wb[18], F(66),
                                     wb[19], F(43), pre);
  k_lstm<<<1, 128, 0, stream>>>(pre, F(44), hsb);
  k_head<<<N_ / 16, 128, 0, stream>>>(hsb, wb[20], F(62), wb[21], F(54),
                                      wb[22], F(56), conc);
  k_bcast<<<(T_ * N_ + 255) / 256, 256, 0, stream>>>(conc, (float*)d_out);
#undef F
}
